// SpatialSpectralAttention_9045201126118
// MI455X (gfx1250) — matrix operand order repaired and checked by reading, not compiled, not run on hardware
//
#include <hip/hip_runtime.h>
#include <hip/hip_bf16.h>

// ---------------- problem constants ----------------
constexpr int BN_   = 8;     // batch
constexpr int CC    = 512;   // channels
constexpr int HW    = 1024;  // 32*32
constexpr int MID   = 64;    // 512/8

// ---------------- WMMA plumbing (gfx1250, wave32) ----------------
typedef __attribute__((ext_vector_type(16))) __bf16          v16bf;
typedef __attribute__((ext_vector_type(16))) unsigned short  v16u;
typedef __attribute__((ext_vector_type(8)))  float           v8f;

__device__ __forceinline__ unsigned short f32_to_bf16(float f) {
  unsigned int u = __float_as_uint(f);
  u += 0x7FFFu + ((u >> 16) & 1u);   // round-to-nearest-even
  return (unsigned short)(u >> 16);
}

__device__ __forceinline__ v8f wmma_bf16(v16u a, v16u b, v8f c) {
  return __builtin_amdgcn_wmma_f32_16x16x32_bf16(
      false, __builtin_bit_cast(v16bf, a),
      false, __builtin_bit_cast(v16bf, b),
      (short)0, c, false, false);
}

// 16-bit A (16x32 MxK): lane = hf*16+lr holds row M=lr; element e: K = hf*16 + e
__device__ __forceinline__ int a_kof(int e, int hf) {
  return hf * 16 + e;
}
// 16-bit B (32x16 KxN): lane holds column N=lr; element e: K = hf*16 + e
__device__ __forceinline__ int b_kof(int e, int hf) { return hf * 16 + e; }

// ---------------- gfx1250 async copy to LDS (ASYNCcnt path) ----------------
// offset: is added to both the LDS address (vdst) and the global address (vaddr)
__device__ __forceinline__ void async_ld_b128x2(unsigned ldsoff, const void* g) {
  unsigned long long ga = (unsigned long long)(uintptr_t)g;
  asm volatile("global_load_async_to_lds_b128 %0, %1, off"
               :: "v"(ldsoff), "v"(ga) : "memory");
  asm volatile("global_load_async_to_lds_b128 %0, %1, off offset:16"
               :: "v"(ldsoff), "v"(ga) : "memory");
}
__device__ __forceinline__ void wait_async0() {
  asm volatile("s_wait_asynccnt 0x0" ::: "memory");
}

// ============================================================
// K0a: x (f32, [b][c][n]) -> bf16 TRANSPOSED [b][n][c]; done once so the conv's
// B-side (and its LDS slab) is K-major => contiguous per-lane fragment reads.
// ============================================================
__global__ void ssa_cvt_x(const float* __restrict__ x, unsigned short* __restrict__ xbfT) {
  int i = blockIdx.x * blockDim.x + threadIdx.x;   // 4,194,304 elements
  int b = i >> 19;
  int r = i & 524287;
  int c = r >> 10;
  int n = r & 1023;
  xbfT[((size_t)b * HW + n) * CC + c] = f32_to_bf16(x[i]);
}

// K0b: weights -> bf16 (Wf|Wg|Wh packed, then Wv); A-side, already K-contiguous.
__global__ void ssa_cvt_w(const float* __restrict__ Wf, const float* __restrict__ Wg,
                          const float* __restrict__ Wh, const float* __restrict__ Wv,
                          unsigned short* __restrict__ Wbf3, unsigned short* __restrict__ Wvbf) {
  int i = blockIdx.x * blockDim.x + threadIdx.x;   // 131072
  if (i < 3 * MID * CC) {
    int set = i >> 15, r = i & 32767;
    const float* W = (set == 0) ? Wf : (set == 1) ? Wg : Wh;
    Wbf3[i] = f32_to_bf16(W[r]);
  } else {
    int r = i - 3 * MID * CC;
    Wvbf[r] = f32_to_bf16(Wv[r]);
  }
}

// ============================================================
// K1: pre-BN convs.  out[set][b][o][n] = sum_c W[o,c]*x[b,c,n] + bias[o]
// Block = 256 thr = 8 waves -> 64(M) x 128(N) tile for one (set,b).
// x slab [128 n][32 k] bf16 (rows padded to 40 ushorts) double-buffered in LDS
// via global_load_async_to_lds_b128; B fragments are per-lane contiguous
// (2x ds_load_b128).  Each wave reuses its A fragment across 4 WMMAs.
// grid: 3 sets * 8 b * 8 ntiles = 192 blocks.
// ============================================================
__global__ __launch_bounds__(256) void ssa_conv3(
    const unsigned short* __restrict__ xbfT,
    const unsigned short* __restrict__ Wbf3,
    const float* __restrict__ bfp, const float* __restrict__ bgp, const float* __restrict__ bhp,
    float* __restrict__ Fpre, float* __restrict__ Gpre,
    unsigned short* __restrict__ Hbf) {
  constexpr int RS = 40;                       // padded row stride (ushorts), 80B: 16B-aligned rows
  __shared__ unsigned short tile[2][128 * RS]; // 2 x 10 KB

  int blk = blockIdx.x;
  int set = blk >> 6;              // 0..2
  int rem = blk & 63;
  int b     = rem >> 3;
  int n0blk = (rem & 7) * 128;

  int tid = threadIdx.x;
  int wv = tid >> 5, lane = tid & 31;
  int hfh = lane >> 4, lr = lane & 15;
  int mt = wv & 3, nh = wv >> 2;   // wave: M-tile 0..3, N-half 0..1
  int o0 = mt * 16;

  const unsigned short* Wset = Wbf3 + (size_t)set * MID * CC;
  const float* bias = (set == 0) ? bfp : (set == 1) ? bgp : bhp;

  // staging: thread t copies 32B of row n = t/2 (k-contiguous), half = t&1
  int sn = tid >> 1;               // 0..127
  int shalf = tid & 1;
  const unsigned short* gsrc =
      xbfT + ((size_t)b * HW + n0blk + sn) * CC + shalf * 16;
  unsigned ldsoff[2] = {
    (unsigned)(uintptr_t)&tile[0][sn * RS + shalf * 16],
    (unsigned)(uintptr_t)&tile[1][sn * RS + shalf * 16]
  };

  async_ld_b128x2(ldsoff[0], gsrc);            // prefetch slab k0 = 0
  v8f acc[4] = {};
  for (int i = 0; i < CC / 32; ++i) {
    int k0 = i * 32;
    int cur = i & 1;
    wait_async0();                 // our slab writes done
    __syncthreads();               // everyone's writes done + prev reads of other buf done
    if (i < CC / 32 - 1)
      async_ld_b128x2(ldsoff[cur ^ 1], gsrc + (size_t)(k0 + 32));

    v16u au;
#pragma unroll
    for (int e = 0; e < 16; ++e)
      au[e] = Wset[(size_t)(o0 + lr) * CC + k0 + a_kof(e, hfh)];

    const unsigned short* tl = &tile[cur][0];
#pragma unroll
    for (int nt = 0; nt < 4; ++nt) {
      int ncol = nh * 64 + nt * 16 + lr;
      v16u bu;
#pragma unroll
      for (int e = 0; e < 16; ++e)
        bu[e] = tl[ncol * RS + hfh * 16 + e];  // contiguous: 2x ds_load_b128
      acc[nt] = wmma_bf16(au, bu, acc[nt]);
    }
  }

#pragma unroll
  for (int nt = 0; nt < 4; ++nt) {
#pragma unroll
    for (int r = 0; r < 8; ++r) {
      int m = o0 + r + hfh * 8;
      int col = n0blk + nh * 64 + nt * 16 + lr;
      float v = acc[nt][r] + bias[m];
      if (set == 2) {
        Hbf[((size_t)b * MID + m) * HW + col] = f32_to_bf16(v);
      } else {
        float* o = (set == 0) ? Fpre : Gpre;
        o[((size_t)b * MID + m) * HW + col] = v;
      }
    }
  }
}

// ============================================================
// K2: BN batch stats over (B,HW) per channel for Fpre/Gpre. 128 blocks.
// ============================================================
__global__ void ssa_bnstats(const float* __restrict__ F, const float* __restrict__ G,
                            float* __restrict__ stats) {
  int job = blockIdx.x;
  int tensor = job >> 6, ch = job & 63;
  const float* data = tensor ? G : F;
  float s = 0.f, s2 = 0.f;
  for (int i = threadIdx.x; i < BN_ * HW; i += blockDim.x) {
    int b = i >> 10, n = i & 1023;
    float v = data[((size_t)b * MID + ch) * HW + n];
    s += v; s2 += v * v;
  }
  __shared__ float rs[256], rq[256];
  rs[threadIdx.x] = s; rq[threadIdx.x] = s2;
  __syncthreads();
  for (int off = 128; off > 0; off >>= 1) {
    if (threadIdx.x < off) {
      rs[threadIdx.x] += rs[threadIdx.x + off];
      rq[threadIdx.x] += rq[threadIdx.x + off];
    }
    __syncthreads();
  }
  if (threadIdx.x == 0) {
    const float inv_n = 1.0f / (BN_ * HW);
    float mu  = rs[0] * inv_n;
    float var = rq[0] * inv_n - mu * mu;
    stats[job * 2 + 0] = mu;
    stats[job * 2 + 1] = rsqrtf(var + 1e-5f);
  }
}

// ============================================================
// K3: BN apply + ReLU, f32 in -> bf16 out.  F additionally written transposed
// ([b][n][c]) so K6's B-side loads are contiguous.
// ============================================================
__global__ void ssa_bnapply(const float* __restrict__ Fpre, const float* __restrict__ Gpre,
                            const float* __restrict__ stats,
                            const float* __restrict__ gamma_f, const float* __restrict__ beta_f,
                            const float* __restrict__ gamma_g, const float* __restrict__ beta_g,
                            unsigned short* __restrict__ Fbf, unsigned short* __restrict__ Gbf,
                            unsigned short* __restrict__ FbfT) {
  int idx = blockIdx.x * blockDim.x + threadIdx.x;   // 2^20
  int tensor = idx >> 19;
  int r = idx & 524287;          // [b][ch][n]
  int b  = r >> 16;
  int ch = (r >> 10) & 63;
  int n  = r & 1023;
  const float* src = tensor ? Gpre : Fpre;
  const float* gam = tensor ? gamma_g : gamma_f;
  const float* bet = tensor ? beta_g  : beta_f;
  float mu  = stats[(tensor * 64 + ch) * 2 + 0];
  float inv = stats[(tensor * 64 + ch) * 2 + 1];
  float v = (src[r] - mu) * inv * gam[ch] + bet[ch];
  unsigned short bf = f32_to_bf16(v > 0.f ? v : 0.f);
  if (tensor) {
    Gbf[r] = bf;
  } else {
    Fbf[r] = bf;
    FbfT[((size_t)b * HW + n) * MID + ch] = bf;
  }
}

// ============================================================
// K4: 64x64 GEMMs (K=1024): A1 = F F^T, A2 = G G^T, P = H G^T per batch.
// out[i][j] = sum_n X[i,n]*Y[j,n].  Both operands contiguous in K.  384 waves.
// ============================================================
__global__ void ssa_gram(const unsigned short* __restrict__ Fbf,
                         const unsigned short* __restrict__ Gbf,
                         const unsigned short* __restrict__ Hbf,
                         float* __restrict__ A1, float* __restrict__ A2,
                         float* __restrict__ P) {
  int wave = (blockIdx.x * blockDim.x + threadIdx.x) >> 5;
  int lane = threadIdx.x & 31;
  int hfh = lane >> 4, lr = lane & 15;

  int set = wave / (BN_ * 16);
  int rem = wave % (BN_ * 16);
  int b  = rem / 16;
  int mt = (rem >> 2) & 3, nt = rem & 3;

  const unsigned short* X = (set == 0) ? Fbf : (set == 1) ? Gbf : Hbf;
  const unsigned short* Y = (set == 0) ? Fbf : Gbf;
  float*                O = (set == 0) ? A1  : (set == 1) ? A2  : P;

  const unsigned short* Xb = X + (size_t)b * MID * HW;
  const unsigned short* Yb = Y + (size_t)b * MID * HW;
  int m0 = mt * 16, n0 = nt * 16;

  v8f acc = {};
  for (int k0 = 0; k0 < HW; k0 += 32) {
    v16u au, bu;
#pragma unroll
    for (int e = 0; e < 16; ++e) {
      au[e] = Xb[(size_t)(m0 + lr) * HW + k0 + a_kof(e, hfh)];
      bu[e] = Yb[(size_t)(n0 + lr) * HW + k0 + b_kof(e, hfh)];
    }
    acc = wmma_bf16(au, bu, acc);
  }
  float* Ob = O + (size_t)b * MID * MID;
#pragma unroll
  for (int r = 0; r < 8; ++r)
    Ob[(size_t)(m0 + r + hfh * 8) * MID + n0 + lr] = acc[r];
}

// ============================================================
// K5: power iteration on A1*A2 (64x64): lambda_max = sigma_max(z)^2.
// ============================================================
__global__ void ssa_sigma(const float* __restrict__ A1, const float* __restrict__ A2,
                          float* __restrict__ inv_sigma) {
  int b = blockIdx.x;
  const float* M1 = A1 + (size_t)b * MID * MID;
  const float* M2 = A2 + (size_t)b * MID * MID;
  __shared__ float v[64], w[64], red[64];
  int t = threadIdx.x;
  v[t] = 1.0f;
  __syncthreads();
  float lam = 1.0f;
  for (int it = 0; it < 24; ++it) {
    float a = 0.f;
    for (int j = 0; j < 64; ++j) a += M2[t * 64 + j] * v[j];
    w[t] = a;
    __syncthreads();
    float y = 0.f;
    for (int j = 0; j < 64; ++j) y += M1[t * 64 + j] * w[j];
    red[t] = y * y;
    __syncthreads();
    for (int off = 32; off > 0; off >>= 1) {
      if (t < off) red[t] += red[t + off];
      __syncthreads();
    }
    lam = sqrtf(red[0] + 1e-30f);
    __syncthreads();
    v[t] = y / lam;
    __syncthreads();
  }
  if (t == 0) inv_sigma[b] = 1.0f / sqrtf(lam);
}

// ============================================================
// K6: z2 = (1/sigma) * P * F   (M=64, N=1024, K=64).
// B from FbfT ([n][c], contiguous); result stored TRANSPOSED Z2T[n][m]
// (per-lane 8 contiguous ushorts -> b128 stores) for K7's B-side.  2048 waves.
// ============================================================
__global__ void ssa_z2(const float* __restrict__ P, const unsigned short* __restrict__ FbfT,
                       const float* __restrict__ inv_sigma, unsigned short* __restrict__ Z2T) {
  int wave = (blockIdx.x * blockDim.x + threadIdx.x) >> 5;
  int lane = threadIdx.x & 31;
  int hfh = lane >> 4, lr = lane & 15;

  int b   = wave / 256;
  int rem = wave % 256;
  int mt = rem / 64, nt = rem & 63;
  int m0 = mt * 16, n0 = nt * 16;

  const float*          Pb = P    + (size_t)b * MID * MID;
  const unsigned short* Fb = FbfT + (size_t)b * HW * MID;
  float invs = inv_sigma[b];

  v8f acc = {};
  for (int k0 = 0; k0 < MID; k0 += 32) {
    v16u au, bu;
#pragma unroll
    for (int e = 0; e < 16; ++e) {
      au[e] = f32_to_bf16(Pb[(size_t)(m0 + lr) * MID + k0 + a_kof(e, hfh)]);
      bu[e] = Fb[(size_t)(n0 + lr) * MID + k0 + b_kof(e, hfh)];   // contiguous
    }
    acc = wmma_bf16(au, bu, acc);
  }
  unsigned short* Ob = Z2T + (size_t)b * HW * MID;
#pragma unroll
  for (int r = 0; r < 8; ++r)
    Ob[(size_t)(n0 + lr) * MID + m0 + hfh * 8 + r] = f32_to_bf16(acc[r] * invs);
}

// ============================================================
// K7: out = Wv * z2 + bv + x   (M=512, N=1024, K=64).  B from Z2T (contiguous).
// 16384 waves.
// ============================================================
__global__ void ssa_final(const unsigned short* __restrict__ Wvbf, const float* __restrict__ bvp,
                          const float* __restrict__ x, const unsigned short* __restrict__ Z2T,
                          float* __restrict__ out) {
  int wave = (blockIdx.x * blockDim.x + threadIdx.x) >> 5;
  int lane = threadIdx.x & 31;
  int hfh = lane >> 4, lr = lane & 15;

  int b   = wave / 2048;
  int rem = wave % 2048;
  int mt  = rem / 64, nt = rem & 63;
  int m0 = mt * 16, n0 = nt * 16;

  const unsigned short* Zb = Z2T + (size_t)b * HW * MID;

  v8f acc = {};
  for (int k0 = 0; k0 < MID; k0 += 32) {
    v16u au, bu;
#pragma unroll
    for (int e = 0; e < 16; ++e) {
      au[e] = Wvbf[(size_t)(m0 + lr) * MID + k0 + a_kof(e, hfh)];
      bu[e] = Zb[(size_t)(n0 + lr) * MID + k0 + b_kof(e, hfh)];   // contiguous
    }
    acc = wmma_bf16(au, bu, acc);
  }
  const float* xb = x   + (size_t)b * CC * HW;
  float*       ob = out + (size_t)b * CC * HW;
#pragma unroll
  for (int r = 0; r < 8; ++r) {
    int m = m0 + r + hfh * 8;
    size_t off = (size_t)m * HW + n0 + lr;
    ob[off] = acc[r] + bvp[m] + xb[off];
  }
}

// ============================================================
// launch
// ============================================================
extern "C" void kernel_launch(void* const* d_in, const int* in_sizes, int n_in,
                              void* d_out, int out_size, void* d_ws, size_t ws_size,
                              hipStream_t stream) {
  (void)in_sizes; (void)n_in; (void)out_size; (void)ws_size;

  const float* x       = (const float*)d_in[0];
  const float* Wf      = (const float*)d_in[1];
  const float* bf_     = (const float*)d_in[2];
  const float* gamma_f = (const float*)d_in[3];
  const float* beta_f  = (const float*)d_in[4];
  const float* Wg      = (const float*)d_in[5];
  const float* bg_     = (const float*)d_in[6];
  const float* gamma_g = (const float*)d_in[7];
  const float* beta_g  = (const float*)d_in[8];
  const float* Wh      = (const float*)d_in[9];
  const float* bh_     = (const float*)d_in[10];
  const float* Wv      = (const float*)d_in[11];
  const float* bv_     = (const float*)d_in[12];
  float*       out     = (float*)d_out;

  // ---------------- workspace layout ----------------
  char* wsb = (char*)d_ws;
  constexpr size_t SZ_ACT = (size_t)BN_ * MID * HW;           // 524288 elems
  size_t off = 0;
  auto take = [&](size_t bytes) { char* p = wsb + off; off = (off + bytes + 255) & ~(size_t)255; return p; };

  float* Fpre  = (float*)take(SZ_ACT * 4);
  float* Gpre  = (float*)take(SZ_ACT * 4);
  float* STATS = (float*)take(256 * 4);
  float* A1    = (float*)take((size_t)BN_ * MID * MID * 4);
  float* A2    = (float*)take((size_t)BN_ * MID * MID * 4);
  float* P     = (float*)take((size_t)BN_ * MID * MID * 4);
  float* INVS  = (float*)take(8 * 4);
  unsigned short* xbfT = (unsigned short*)take((size_t)BN_ * CC * HW * 2);
  unsigned short* Wbf3 = (unsigned short*)take((size_t)3 * MID * CC * 2);
  unsigned short* Wvbf = (unsigned short*)take((size_t)CC * MID * 2);
  unsigned short* Fbf  = (unsigned short*)take(SZ_ACT * 2);
  unsigned short* Gbf  = (unsigned short*)take(SZ_ACT * 2);
  unsigned short* Hbf  = (unsigned short*)take(SZ_ACT * 2);
  unsigned short* FbfT = (unsigned short*)take(SZ_ACT * 2);
  unsigned short* Z2T  = (unsigned short*)take(SZ_ACT * 2);

  ssa_cvt_x  <<<16384, 256, 0, stream>>>(x, xbfT);
  ssa_cvt_w  <<<512,   256, 0, stream>>>(Wf, Wg, Wh, Wv, Wbf3, Wvbf);
  ssa_conv3  <<<192,   256, 0, stream>>>(xbfT, Wbf3, bf_, bg_, bh_, Fpre, Gpre, Hbf);
  ssa_bnstats<<<128,   256, 0, stream>>>(Fpre, Gpre, STATS);
  ssa_bnapply<<<4096,  256, 0, stream>>>(Fpre, Gpre, STATS, gamma_f, beta_f, gamma_g, beta_g,
                                         Fbf, Gbf, FbfT);
  ssa_gram   <<<48,    256, 0, stream>>>(Fbf, Gbf, Hbf, A1, A2, P);
  ssa_sigma  <<<8,      64, 0, stream>>>(A1, A2, INVS);
  ssa_z2     <<<256,   256, 0, stream>>>(P, FbfT, INVS, Z2T);
  ssa_final  <<<2048,  256, 0, stream>>>(Wvbf, bv_, x, Z2T, out);
}
